// Gate_50878182588478
// MI455X (gfx1250) — matrix operand order repaired and checked by reading, not compiled, not run on hardware
//
#include <hip/hip_runtime.h>

// ---------------------------------------------------------------------------
// MoE gate for MI455X (gfx1250, wave32, WMMA).
//   logits = x @ W + b            (bf16 WMMA, f32 accumulate)
//   p      = softmax(logits)      (16-lane shuffle reductions)
//   mask   = top-2(p)             (max / second-max)
//   out    = p*mask / (colsum+eps) * capacity   (2-pass: atomics then rescale)
//   loss   = mean(density_proxy * density) * E^2
// ---------------------------------------------------------------------------

typedef __attribute__((ext_vector_type(16))) __bf16 v16bf;
typedef __attribute__((ext_vector_type(8)))  float  v8f;

#define T_TOKENS 16384
#define DIMK     4096
#define NEXP     64
#define KBLK     (DIMK / 32)          // 128 K-steps of 32
#define ROWS_PER_BLOCK 128            // 8 waves x 16 rows
#define WS_B_ELEMS (KBLK * 4 * 512)   // 262144 bf16 fragment elements
#define WS_B_BYTES (WS_B_ELEMS * 2)   // 512 KB

// --- zero the 3x64 accumulators (denom, count, proxy) ----------------------
__global__ __launch_bounds__(256) void zero_accums_kernel(float* __restrict__ acc) {
    int i = threadIdx.x;
    if (i < 192) acc[i] = 0.0f;
}

// --- W [4096,64] f32 row-major  ->  bf16 WMMA B-fragments ------------------
// Fragment f = kb*4 + nb holds B tile (K = kb*32..+31, N = nb*16..+15) in the
// contiguous-K order shared with the A loads: lane L -> column n = nb*16 + (L&15); lane-half (L>>4)
// selects K range 16*(L>>4)..+15; 16 bf16 per lane stored contiguously so the
// GEMM loads each lane's fragment slice with two b128s.
__global__ __launch_bounds__(256) void prep_w_kernel(const float* __restrict__ W,
                                                     __bf16* __restrict__ wsB) {
    int d   = blockIdx.x * 256 + threadIdx.x;   // 0 .. 262143
    int f   = d >> 9;                           // fragment index
    int rem = d & 511;
    int L   = rem >> 4;                         // lane 0..31
    int q   = rem & 15;                         // 2*vgpr + half  -> K offset
    int kb  = f >> 2;
    int nb  = f & 3;
    int n   = nb * 16 + (L & 15);
    int k   = kb * 32 + ((L >> 4) << 4) + q;
    wsB[d] = (__bf16)W[(size_t)k * NEXP + n];
}

// --- main pass: GEMM + softmax + top2 + masked write + partial sums --------
__global__ __launch_bounds__(256) void gate_pass1_kernel(
        const float* __restrict__ x, const float* __restrict__ bias,
        const __bf16* __restrict__ wsB, float* __restrict__ out,
        float* __restrict__ denom, float* __restrict__ csum,
        float* __restrict__ psum) {
    const int lane = threadIdx.x & 31;
    const int wave = threadIdx.x >> 5;
    const int g    = lane >> 4;       // lane-half
    const int ln   = lane & 15;
    const int m0   = blockIdx.x * ROWS_PER_BLOCK + wave * 16;

    const v8f zv = {0.f, 0.f, 0.f, 0.f, 0.f, 0.f, 0.f, 0.f};
    v8f c0 = zv, c1 = zv, c2 = zv, c3 = zv;

    // A (16-bit, 16x32): lane holds row m0+ln, K range [16g,16g+16)
    // of each 32-wide K step -> contiguous loads (matches B packing).
    const float*  arow  = x + (size_t)(m0 + ln) * DIMK + 16 * g;
    const __bf16* bbase = wsB + (size_t)lane * 16;

    for (int kb = 0; kb < KBLK; ++kb) {
        const float* ap = arow + kb * 32;
        __builtin_prefetch(ap + 32, 0, 0);     // next K-step (global_prefetch)
        float4 q0 = ((const float4*)ap)[0];
        float4 q1 = ((const float4*)ap)[1];
        float4 q2 = ((const float4*)(ap + 8))[0];
        float4 q3 = ((const float4*)(ap + 8))[1];
        v16bf a;
        a[0]  = (__bf16)q0.x; a[1]  = (__bf16)q0.y; a[2]  = (__bf16)q0.z; a[3]  = (__bf16)q0.w;
        a[4]  = (__bf16)q1.x; a[5]  = (__bf16)q1.y; a[6]  = (__bf16)q1.z; a[7]  = (__bf16)q1.w;
        a[8]  = (__bf16)q2.x; a[9]  = (__bf16)q2.y; a[10] = (__bf16)q2.z; a[11] = (__bf16)q2.w;
        a[12] = (__bf16)q3.x; a[13] = (__bf16)q3.y; a[14] = (__bf16)q3.z; a[15] = (__bf16)q3.w;

        const __bf16* bp = bbase + (size_t)kb * 2048;   // 4 frags * 512 elems
        v16bf b0 = *(const v16bf*)(bp);
        v16bf b1 = *(const v16bf*)(bp + 512);
        v16bf b2 = *(const v16bf*)(bp + 1024);
        v16bf b3 = *(const v16bf*)(bp + 1536);

        c0 = __builtin_amdgcn_wmma_f32_16x16x32_bf16(false, a, false, b0, (short)0, c0, false, false);
        c1 = __builtin_amdgcn_wmma_f32_16x16x32_bf16(false, a, false, b1, (short)0, c1, false, false);
        c2 = __builtin_amdgcn_wmma_f32_16x16x32_bf16(false, a, false, b2, (short)0, c2, false, false);
        c3 = __builtin_amdgcn_wmma_f32_16x16x32_bf16(false, a, false, b3, (short)0, c3, false, false);
    }

    // C layout: vgpr r, lane -> row m0 + r + 8*g, col nb*16 + ln.
    float bias_v[4];
    #pragma unroll
    for (int nb = 0; nb < 4; ++nb) bias_v[nb] = bias[nb * 16 + ln];

    float dsum[4] = {0.f, 0.f, 0.f, 0.f};
    float cnt[4]  = {0.f, 0.f, 0.f, 0.f};
    float ps[4]   = {0.f, 0.f, 0.f, 0.f};

    #pragma unroll
    for (int r = 0; r < 8; ++r) {
        float l[4] = {c0[r] + bias_v[0], c1[r] + bias_v[1],
                      c2[r] + bias_v[2], c3[r] + bias_v[3]};
        // row max (16 lanes of one half hold the whole 64-wide row)
        float mx = fmaxf(fmaxf(l[0], l[1]), fmaxf(l[2], l[3]));
        #pragma unroll
        for (int off = 1; off < 16; off <<= 1) mx = fmaxf(mx, __shfl_xor(mx, off, 32));
        float e[4]; float s = 0.f;
        #pragma unroll
        for (int nb = 0; nb < 4; ++nb) { e[nb] = __expf(l[nb] - mx); s += e[nb]; }
        #pragma unroll
        for (int off = 1; off < 16; off <<= 1) s += __shfl_xor(s, off, 32);
        float inv = 1.0f / s;
        float p[4];
        #pragma unroll
        for (int nb = 0; nb < 4; ++nb) p[nb] = e[nb] * inv;
        // top-2: global max, then max excluding the max element
        float m1 = fmaxf(fmaxf(p[0], p[1]), fmaxf(p[2], p[3]));
        #pragma unroll
        for (int off = 1; off < 16; off <<= 1) m1 = fmaxf(m1, __shfl_xor(m1, off, 32));
        float m2 = -1.0f;
        #pragma unroll
        for (int nb = 0; nb < 4; ++nb) { float t = (p[nb] == m1) ? -1.0f : p[nb]; m2 = fmaxf(m2, t); }
        #pragma unroll
        for (int off = 1; off < 16; off <<= 1) m2 = fmaxf(m2, __shfl_xor(m2, off, 32));

        const int row  = m0 + r + 8 * g;
        float* orow = out + (size_t)row * NEXP + ln;
        #pragma unroll
        for (int nb = 0; nb < 4; ++nb) {
            float kbit = (p[nb] >= m2) ? 1.0f : 0.0f;   // keeps top-2 (m1 >= m2)
            float mp   = p[nb] * kbit;
            orow[nb * 16] = mp;                          // un-normalized for pass 2
            dsum[nb] += mp; cnt[nb] += kbit; ps[nb] += p[nb];
        }
    }

    // fold the two lane-halves (same column, different rows), then atomics
    #pragma unroll
    for (int nb = 0; nb < 4; ++nb) {
        float dv = dsum[nb] + __shfl_xor(dsum[nb], 16, 32);
        float cv = cnt[nb]  + __shfl_xor(cnt[nb], 16, 32);
        float pv = ps[nb]   + __shfl_xor(ps[nb], 16, 32);
        if (g == 0) {
            atomicAdd(&denom[nb * 16 + ln], dv);
            atomicAdd(&csum[nb * 16 + ln],  cv);
            atomicAdd(&psum[nb * 16 + ln],  pv);
        }
    }
}

// --- pass 2: out *= capacity / (denom + eps) -------------------------------
__global__ __launch_bounds__(256) void rescale_kernel(float* __restrict__ out,
                                                      const float* __restrict__ denom) {
    int i = blockIdx.x * 256 + threadIdx.x;     // exactly TOKENS*NEXP threads
    int e = i & (NEXP - 1);
    out[i] *= 16384.0f / (denom[e] + 1e-6f);
}

// --- loss = mean_e(proxy * density) * E^2 ----------------------------------
__global__ __launch_bounds__(64) void loss_kernel(const float* __restrict__ csum,
                                                  const float* __restrict__ psum,
                                                  float* __restrict__ out_loss) {
    __shared__ float red[64];
    int e = threadIdx.x;
    const float invT = 1.0f / (float)T_TOKENS;
    red[e] = (psum[e] * invT) * (csum[e] * invT);
    __syncthreads();
    for (int s = 32; s > 0; s >>= 1) {
        if (e < s) red[e] += red[e + s];
        __syncthreads();
    }
    if (e == 0) out_loss[0] = red[0] * 64.0f;   // (sum/E) * E^2 = sum * E
}

extern "C" void kernel_launch(void* const* d_in, const int* in_sizes, int n_in,
                              void* d_out, int out_size, void* d_ws, size_t ws_size,
                              hipStream_t stream) {
    (void)in_sizes; (void)n_in; (void)out_size;
    const float* x = (const float*)d_in[0];
    const float* W = (const float*)d_in[1];
    const float* b = (const float*)d_in[2];
    float* out = (float*)d_out;

    if (ws_size < (size_t)WS_B_BYTES + 192 * sizeof(float)) return;
    __bf16* wsB  = (__bf16*)d_ws;
    float*  acc  = (float*)((char*)d_ws + WS_B_BYTES);
    float* denom = acc;
    float* csum  = acc + 64;
    float* psum  = acc + 128;

    zero_accums_kernel<<<1, 256, 0, stream>>>(acc);
    prep_w_kernel<<<(DIMK * NEXP) / 256, 256, 0, stream>>>(W, wsB);
    gate_pass1_kernel<<<T_TOKENS / ROWS_PER_BLOCK, 256, 0, stream>>>(
        x, b, wsB, out, denom, csum, psum);
    rescale_kernel<<<(T_TOKENS * NEXP) / 256, 256, 0, stream>>>(out, denom);
    loss_kernel<<<1, 64, 0, stream>>>(csum, psum, out + (size_t)T_TOKENS * NEXP);
}
